// ClusterKLLoss_16003048145636
// MI455X (gfx1250) — matrix operand order repaired and checked by reading, not compiled, not run on hardware
//
#include <hip/hip_runtime.h>
#include <hip/hip_bf16.h>

#define BATCH 4096
#define DIM   2048
#define INV_T 2.0f   // 1/TEMPERATURE, TEMPERATURE = 0.5

typedef __attribute__((ext_vector_type(16))) __bf16 v16bf;
typedef __attribute__((ext_vector_type(8)))  float  v8f;

union FragBF16 {
    v16bf v;
    uint4 u[2];
};

// ---------------------------------------------------------------------------
// Row softmax preprocessing.
//   compute_hneg == 0: out[row,k] = bf16(log_softmax(x[row])_k)          (Li)
//   compute_hneg == 1: out[row,k] = bf16(softmax(x[row])_k)              (Q)
//                      hneg[row]  = sum_k Q*logQ  (negative entropy)
// One block (256 threads) per row of 2048 elements.
// ---------------------------------------------------------------------------
__global__ void __launch_bounds__(256)
prep_rows(const float* __restrict__ x, __bf16* __restrict__ out,
          float* __restrict__ hneg, int compute_hneg) {
    const int row = blockIdx.x;
    const float* xr = x + (size_t)row * DIM;
    __shared__ float red[256];
    const int t = threadIdx.x;

    // pass 1: row max
    float m = -1e30f;
    for (int k = t; k < DIM; k += 256) m = fmaxf(m, xr[k]);
    red[t] = m; __syncthreads();
    for (int s = 128; s > 0; s >>= 1) {
        if (t < s) red[t] = fmaxf(red[t], red[t + s]);
        __syncthreads();
    }
    m = red[0]; __syncthreads();

    // pass 2: sum of exp
    float sum = 0.f;
    for (int k = t; k < DIM; k += 256) sum += __expf(xr[k] - m);
    red[t] = sum; __syncthreads();
    for (int s = 128; s > 0; s >>= 1) {
        if (t < s) red[t] += red[t + s];
        __syncthreads();
    }
    const float lse = m + __logf(red[0]);
    __syncthreads();

    // pass 3: emit bf16 values (+ negative entropy)
    float h = 0.f;
    for (int k = t; k < DIM; k += 256) {
        const float l = xr[k] - lse;          // log-softmax
        if (compute_hneg) {
            const float q = __expf(l);
            out[(size_t)row * DIM + k] = (__bf16)q;
            h += q * l;
        } else {
            out[(size_t)row * DIM + k] = (__bf16)l;
        }
    }
    if (compute_hneg) {
        red[t] = h; __syncthreads();
        for (int s = 128; s > 0; s >>= 1) {
            if (t < s) red[t] += red[t + s];
            __syncthreads();
        }
        if (t == 0) hneg[row] = red[0];
    }
}

// ---------------------------------------------------------------------------
// bf16 WMMA GEMM: logits[i,j] = (sum_k Li[i,k]*Q[j,k] - hneg[j]) * (1/T)
// Block = 256 threads = 8 waves arranged 2(M) x 4(N); block tile 128x256.
// Wave tile 64x64 = 4x4 accumulators of 16x16 -> 16 WMMAs per K=32 step
// against 16 b128 loads (32 FLOP/byte at the L0/L2 interface).
// Operands are L2-resident (32 MB << 192 MB), so no LDS staging needed.
// ---------------------------------------------------------------------------
__global__ void __launch_bounds__(256)
kl_gemm(const __bf16* __restrict__ Li, const __bf16* __restrict__ Q,
        const float* __restrict__ hneg, float* __restrict__ logits) {
    const int lane  = threadIdx.x & 31;
    const int wave  = threadIdx.x >> 5;
    const int waveM = wave & 1;                     // 0..1
    const int waveN = wave >> 1;                    // 0..3
    const int i0 = blockIdx.y * 128 + waveM * 64;   // output rows for this wave
    const int j0 = blockIdx.x * 256 + waveN * 64;   // output cols for this wave

    const int rA  = lane & 15;            // matrix row within a 16-row tile
    const int hiA = (lane >> 4) << 3;     // A: lanes 16-31 take K offset +8
    const int hiB = (lane >> 4) << 3;     // B: lanes 16-31 take K offset +8

    v8f acc[4][4];
#pragma unroll
    for (int mt = 0; mt < 4; ++mt)
#pragma unroll
        for (int nt = 0; nt < 4; ++nt)
            acc[mt][nt] = (v8f){0.f, 0.f, 0.f, 0.f, 0.f, 0.f, 0.f, 0.f};

    const __bf16* aBase[4];
#pragma unroll
    for (int mt = 0; mt < 4; ++mt)
        aBase[mt] = Li + (size_t)(i0 + mt * 16 + rA) * DIM;
    const __bf16* bBase[4];
#pragma unroll
    for (int nt = 0; nt < 4; ++nt)
        bBase[nt] = Q + (size_t)(j0 + nt * 16 + rA) * DIM;

    for (int kk = 0; kk < DIM; kk += 32) {
        FragBF16 a[4], b[4];
        // A fragment (16x32 bf16): two contiguous 16B chunks per lane.
        //   lanes 0-15 hold K = {kk..kk+7, kk+16..kk+23}, lanes 16-31 get +8.
#pragma unroll
        for (int mt = 0; mt < 4; ++mt) {
            a[mt].u[0] = *(const uint4*)(aBase[mt] + kk + hiA);
            a[mt].u[1] = *(const uint4*)(aBase[mt] + kk + 16 + hiA);
        }
        // B fragment (32x16 bf16, B = Q-tile transposed): lane n = lane&15
        // holds two 16B K-chunks of row (j0+nt*16+n) of Q, same K order as A.
#pragma unroll
        for (int nt = 0; nt < 4; ++nt) {
            b[nt].u[0] = *(const uint4*)(bBase[nt] + kk + hiB);
            b[nt].u[1] = *(const uint4*)(bBase[nt] + kk + 16 + hiB);
        }
#pragma unroll
        for (int mt = 0; mt < 4; ++mt)
#pragma unroll
            for (int nt = 0; nt < 4; ++nt)
                acc[mt][nt] = __builtin_amdgcn_wmma_f32_16x16x32_bf16(
                    false, a[mt].v, false, b[nt].v,
                    (short)0, acc[mt][nt], false, false);
    }

    // Epilogue: C/D layout -> VGPR r, lanes 0-15: M=r, lanes 16-31: M=r+8;
    // N = lane & 15. Fuse the (cross - hneg[j]) * (1/T) transform.
    const int mHi = (lane >> 4) << 3;
#pragma unroll
    for (int nt = 0; nt < 4; ++nt) {
        const int n = j0 + nt * 16 + (lane & 15);
        const float hn = hneg[n];
#pragma unroll
        for (int mt = 0; mt < 4; ++mt) {
#pragma unroll
            for (int r = 0; r < 8; ++r) {
                const int m = i0 + mt * 16 + mHi + r;
                logits[(size_t)m * BATCH + n] = (acc[mt][nt][r] - hn) * INV_T;
            }
        }
    }
}

// ---------------------------------------------------------------------------
// Per-row logsumexp over logits (4096 cols); rowloss[i] = logits[i,i] - lse_i
// ---------------------------------------------------------------------------
__global__ void __launch_bounds__(256)
row_lse(const float* __restrict__ logits, float* __restrict__ rowloss) {
    const int row = blockIdx.x;
    const float* lr = logits + (size_t)row * BATCH;
    __shared__ float red[256];
    const int t = threadIdx.x;

    float m = -1e30f;
    for (int k = t; k < BATCH; k += 256) m = fmaxf(m, lr[k]);
    red[t] = m; __syncthreads();
    for (int s = 128; s > 0; s >>= 1) {
        if (t < s) red[t] = fmaxf(red[t], red[t + s]);
        __syncthreads();
    }
    m = red[0]; __syncthreads();

    float sum = 0.f;
    for (int k = t; k < BATCH; k += 256) sum += __expf(lr[k] - m);
    red[t] = sum; __syncthreads();
    for (int s = 128; s > 0; s >>= 1) {
        if (t < s) red[t] += red[t + s];
        __syncthreads();
    }
    if (t == 0) rowloss[row] = lr[row] - (m + __logf(red[0]));
}

// ---------------------------------------------------------------------------
// loss = -sum_i rowloss[i] / BATCH   (single block, deterministic)
// ---------------------------------------------------------------------------
__global__ void __launch_bounds__(256)
final_loss(const float* __restrict__ rowloss, float* __restrict__ out) {
    __shared__ float red[256];
    const int t = threadIdx.x;
    float s = 0.f;
    for (int k = t; k < BATCH; k += 256) s += rowloss[k];
    red[t] = s; __syncthreads();
    for (int st = 128; st > 0; st >>= 1) {
        if (t < st) red[t] += red[t + st];
        __syncthreads();
    }
    if (t == 0) out[0] = -red[0] / (float)BATCH;
}

// ---------------------------------------------------------------------------
extern "C" void kernel_launch(void* const* d_in, const int* in_sizes, int n_in,
                              void* d_out, int out_size, void* d_ws, size_t ws_size,
                              hipStream_t stream) {
    (void)in_sizes; (void)n_in; (void)out_size; (void)ws_size;
    const float* c_i = (const float*)d_in[0];
    const float* c_j = (const float*)d_in[1];

    char* ws = (char*)d_ws;
    __bf16* Li      = (__bf16*)(ws);                                   // 16 MiB
    __bf16* Q       = (__bf16*)(ws + (size_t)16 * 1024 * 1024);        // 16 MiB
    float*  logits  = (float*)(ws + (size_t)32 * 1024 * 1024);         // 64 MiB
    float*  hneg    = (float*)(ws + (size_t)96 * 1024 * 1024);         // 16 KiB
    float*  rowloss = (float*)(ws + (size_t)96 * 1024 * 1024 + 16384); // 16 KiB

    prep_rows<<<BATCH, 256, 0, stream>>>(c_i, Li, hneg, 0);  // hneg unused here
    prep_rows<<<BATCH, 256, 0, stream>>>(c_j, Q,  hneg, 1);

    dim3 grid(BATCH / 256, BATCH / 128);   // (N tiles, M tiles)
    kl_gemm<<<grid, 256, 0, stream>>>(Li, Q, hneg, logits);

    row_lse<<<BATCH, 256, 0, stream>>>(logits, rowloss);
    final_loss<<<1, 256, 0, stream>>>(rowloss, (float*)d_out);
}
